// VectorQuantizerEMA_87368224735733
// MI455X (gfx1250) — matrix operand order repaired and checked by reading, not compiled, not run on hardware
//
#include <hip/hip_runtime.h>
#include <hip/hip_bf16.h>

typedef __attribute__((ext_vector_type(16))) _Float16 v16h;
typedef __attribute__((ext_vector_type(8)))  float    v8f;

#define NUM_EMB 1024
#define DIM     128
#define HW      1024
#define NROWS   65536
#define DECAYF  0.99f
#define OMDECAY 0.01f
#define EPSF    1e-5f
#define CCOEF   0.25f

// d_out layout (floats), reference tuple order:
#define OFF_Q    0
#define OFF_IDX  8388608
#define OFF_LOSS 8454144
#define OFF_EMB  8454145
#define OFF_NECS 8585217
#define OFF_EMAW 8586241

// workspace byte offsets
#define WS_DW     0         // float[131072]
#define WS_COUNTS 524288    // float[1024]
#define WS_LOSS   528384    // float[1]
#define WS_ENORM  528448    // float[1024]
#define WS_EH     532544    // _Float16[131072]
#define WS_EL     794688    // _Float16[131072]

// dynamic LDS layout (bytes) for vq_main
#define LDS_AS    0         // float[128*128]  x-tile, As[d*128 + r]
#define LDS_EH    65536     // unsigned[2][16*68]  hi halves, double-buffered, stride-68 pad
#define LDS_EL    74240     // unsigned[2][16*68]  lo halves
#define LDS_IDX   82944     // int[128]
#define LDS_RED   83456     // float[256]
#define LDS_TOTAL 84480

// CDNA5 async DMA: memory -> LDS, per-lane 16B, tracked by ASYNCcnt
__device__ __forceinline__ void async_b128(unsigned lds_byte, unsigned goff_byte,
                                           unsigned long long sbase) {
  asm volatile("global_load_async_to_lds_b128 %0, %1, %2"
               :: "v"(lds_byte), "v"(goff_byte), "s"(sbase) : "memory");
}
__device__ __forceinline__ void wait_async0() {
  asm volatile("s_wait_asynccnt 0x0" ::: "memory");
}

// ---------------- prep: split embed into f16 hi/lo, norms, zero accumulators --------
__global__ __launch_bounds__(128)
void vq_prep(const float* __restrict__ embed,
             float* __restrict__ dw, float* __restrict__ counts,
             float* __restrict__ lossws, float* __restrict__ enorm,
             _Float16* __restrict__ eh, _Float16* __restrict__ el) {
  const int k = blockIdx.x, t = threadIdx.x;
  const int i = k * DIM + t;
  float e = embed[i];
  _Float16 h = (_Float16)e;
  eh[i] = h;
  el[i] = (_Float16)(e - (float)h);
  dw[i] = 0.0f;
  __shared__ float red[DIM];
  red[t] = e * e;
  __syncthreads();
  for (int s = 64; s > 0; s >>= 1) { if (t < s) red[t] += red[t + s]; __syncthreads(); }
  if (t == 0) { enorm[k] = red[0]; counts[k] = 0.0f; if (k == 0) lossws[0] = 0.0f; }
}

// ---------------- main: distances via split-f16 WMMA + async-DMA staging -----------
__global__ __launch_bounds__(256)
void vq_main(const float* __restrict__ z_e, const float* __restrict__ embed,
             const unsigned* __restrict__ ehg, const unsigned* __restrict__ elg,
             const float* __restrict__ enorm,
             float* __restrict__ out_q, float* __restrict__ out_idx,
             float* __restrict__ counts, float* __restrict__ dw,
             float* __restrict__ lossws) {
  extern __shared__ char smem[];
  float*    As    = (float*)(smem + LDS_AS);
  unsigned* EhB   = (unsigned*)(smem + LDS_EH);
  unsigned* ElB   = (unsigned*)(smem + LDS_EL);
  int*      wgIdx = (int*)(smem + LDS_IDX);
  float*    red   = (float*)(smem + LDS_RED);
  const unsigned lds0 = (unsigned)(size_t)smem;   // low 32 bits = LDS byte offset

  const int tid  = threadIdx.x;
  const int wave = tid >> 5;
  const int lane = tid & 31;
  const int hf   = lane >> 4;
  const int lrow = lane & 15;
  const int n0   = blockIdx.x * 128;   // 128 rows per WG (8 waves x 16 rows)
  const int b    = n0 >> 10;
  const int hw0  = n0 & 1023;
  const long zb  = (long)b * (DIM * HW) + hw0;

  const unsigned long long zbase  = (unsigned long long)(z_e + zb);
  const unsigned long long ehbase = (unsigned long long)ehg;
  const unsigned long long elbase = (unsigned long long)elg;

  // async-DMA the 128x128 f32 x-tile into LDS (4096 x b128, coalesced along hw)
  for (int s = tid; s < 4096; s += 256) {
    int d = s >> 5, rr = (s & 31) << 2;
    async_b128(lds0 + LDS_AS + ((unsigned)(d * 128 + rr) << 2),
               ((unsigned)(d * HW + rr)) << 2, zbase);
  }
  // async-DMA code tile 0 into buffer 0 (one b128 slot per thread per plane)
  {
    int code = tid >> 4, u4 = (tid & 15) << 2;
    unsigned goff = ((unsigned)(code * 64 + u4)) << 2;
    unsigned loff = ((unsigned)(code * 68 + u4)) << 2;
    async_b128(lds0 + LDS_EH + loff, goff, ehbase);
    async_b128(lds0 + LDS_EL + loff, goff, elbase);
  }
  wait_async0();
  __syncthreads();

  union V16 { v16h v; _Float16 h[16]; unsigned u[8]; uint4 q[2]; };
  union V8F { v8f v; float f[8]; };

  // A fragments (16x32 f16 layout), split f32 -> hi/lo f16
  V16 AH[4], AL[4];
  const int mb = wave * 16 + lrow;
  #pragma unroll
  for (int c = 0; c < 4; ++c) {
    #pragma unroll
    for (int i = 0; i < 8; ++i) {
      int bk = 2 * i + (hf ? 16 : 0);
      int dd = c * 32 + bk;
      float x0 = As[dd * 128 + mb];
      float x1 = As[(dd + 1) * 128 + mb];
      _Float16 h0 = (_Float16)x0, h1 = (_Float16)x1;
      AH[c].h[2 * i]     = h0; AL[c].h[2 * i]     = (_Float16)(x0 - (float)h0);
      AH[c].h[2 * i + 1] = h1; AL[c].h[2 * i + 1] = (_Float16)(x1 - (float)h1);
    }
  }

  float bestv[8]; int besti[8];
  #pragma unroll
  for (int r = 0; r < 8; ++r) { bestv[r] = 3.4e38f; besti[r] = 0; }

  for (int t = 0; t < NUM_EMB / 16; ++t) {
    const int bb = t & 1;
    if (t + 1 < NUM_EMB / 16) {   // prefetch next tile into the other buffer via async DMA
      int k0n = (t + 1) * 16;
      int code = tid >> 4, u4 = (tid & 15) << 2;
      unsigned goff = ((unsigned)((k0n + code) * 64 + u4)) << 2;
      unsigned loff = ((unsigned)(((t + 1) & 1) * 1088 + code * 68 + u4)) << 2;
      async_b128(lds0 + LDS_EH + loff, goff, ehbase);
      async_b128(lds0 + LDS_EL + loff, goff, elbase);
    }

    const unsigned* Eh = EhB + bb * 1088;
    const unsigned* El = ElB + bb * 1088;
    // Three independent accumulator chains -> no WMMA->WMMA RAW interlock stalls.
    V8F aHH = {}, aHL = {}, aLH = {};
    #pragma unroll
    for (int c = 0; c < 4; ++c) {
      // per-lane fragment slice is 32 contiguous bytes, 16B-aligned (272=17*16),
      // b128 reads tile all 64 banks across 16 lanes -> conflict-free ds_load_b128
      const uint4* Eh4 = (const uint4*)(Eh + lrow * 68 + c * 16 + hf * 8);
      const uint4* El4 = (const uint4*)(El + lrow * 68 + c * 16 + hf * 8);
      V16 BH, BL;
      BH.q[0] = Eh4[0]; BH.q[1] = Eh4[1];
      BL.q[0] = El4[0]; BL.q[1] = El4[1];
      aHH.v = __builtin_amdgcn_wmma_f32_16x16x32_f16(false, AH[c].v, false, BH.v, (short)0, aHH.v, false, false);
      aHL.v = __builtin_amdgcn_wmma_f32_16x16x32_f16(false, AH[c].v, false, BL.v, (short)0, aHL.v, false, false);
      aLH.v = __builtin_amdgcn_wmma_f32_16x16x32_f16(false, AL[c].v, false, BH.v, (short)0, aLH.v, false, false);
    }

    const float en = enorm[t * 16 + lrow];
    const int code = t * 16 + lrow;
    #pragma unroll
    for (int r = 0; r < 8; ++r) {
      float dot = (aHH.f[r] + aHL.f[r]) + aLH.f[r];
      float s = __builtin_fmaf(-2.0f, dot, en);
      if (s < bestv[r]) { bestv[r] = s; besti[r] = code; }
    }

    wait_async0();      // next tile landed in LDS
    __syncthreads();    // all waves done reading current buffer
  }

  // min-reduce across the 16 code-column lanes (xor masks stay within 16-lane halves)
  #pragma unroll
  for (int m = 1; m < 16; m <<= 1) {
    #pragma unroll
    for (int r = 0; r < 8; ++r) {
      float ov = __shfl_xor(bestv[r], m, 32);
      int   oi = __shfl_xor(besti[r], m, 32);
      if (ov < bestv[r] || (ov == bestv[r] && oi < besti[r])) { bestv[r] = ov; besti[r] = oi; }
    }
  }
  if (lrow == 0) {
    #pragma unroll
    for (int r = 0; r < 8; ++r) wgIdx[wave * 16 + hf * 8 + r] = besti[r];
  }
  __syncthreads();

  if (tid < 128) {
    int k = wgIdx[tid];
    out_idx[n0 + tid] = (float)k;
    atomicAdd(&counts[k], 1.0f);
  }

  // vectorized post-phase: 4 consecutive rows per thread -> b128 LDS reads + b128 stores
  float lacc = 0.0f;
  for (int i0 = tid * 4; i0 < DIM * 128; i0 += 1024) {
    int d = i0 >> 7, r = i0 & 127;
    float4 x = *(const float4*)(As + i0);
    int k0 = wgIdx[r], k1 = wgIdx[r + 1], k2 = wgIdx[r + 2], k3 = wgIdx[r + 3];
    float4 q;
    q.x = embed[k0 * DIM + d];
    q.y = embed[k1 * DIM + d];
    q.z = embed[k2 * DIM + d];
    q.w = embed[k3 * DIM + d];
    float d0 = q.x - x.x, d1 = q.y - x.y, d2 = q.z - x.z, d3 = q.w - x.w;
    lacc += d0 * d0 + d1 * d1 + d2 * d2 + d3 * d3;
    *(float4*)(out_q + zb + (long)d * HW + r) = q;   // 16B-aligned, coalesced
    atomicAdd(&dw[k0 * DIM + d], x.x);
    atomicAdd(&dw[k1 * DIM + d], x.y);
    atomicAdd(&dw[k2 * DIM + d], x.z);
    atomicAdd(&dw[k3 * DIM + d], x.w);
  }
  red[tid] = lacc;
  __syncthreads();
  for (int s = 128; s > 0; s >>= 1) { if (tid < s) red[tid] += red[tid + s]; __syncthreads(); }
  if (tid == 0) atomicAdd(lossws, red[0]);
}

// ---------------- finalize: EMA updates, normalization, loss -----------------------
__global__ __launch_bounds__(1024)
void vq_finalize(const float* __restrict__ ema_cs, const float* __restrict__ ema_w,
                 const float* __restrict__ counts, const float* __restrict__ dw,
                 const float* __restrict__ lossws, float* __restrict__ out) {
  __shared__ float arr[NUM_EMB];
  const int k = threadIdx.x;
  float csn = DECAYF * ema_cs[k] + OMDECAY * counts[k];
  out[OFF_NECS + k] = csn;
  arr[k] = csn;
  __syncthreads();
  for (int s = 512; s > 0; s >>= 1) { if (k < s) arr[k] += arr[k + s]; __syncthreads(); }
  float n = arr[0];
  float cs = (csn + EPSF) / (n + (float)NUM_EMB * EPSF) * n;
  for (int d = 0; d < DIM; ++d) {
    float wn = DECAYF * ema_w[k * DIM + d] + OMDECAY * dw[k * DIM + d];
    out[OFF_EMAW + k * DIM + d] = wn;
    out[OFF_EMB  + k * DIM + d] = wn / cs;
  }
  if (k == 0) out[OFF_LOSS] = lossws[0] * (CCOEF / (float)(NROWS * DIM));
}

extern "C" void kernel_launch(void* const* d_in, const int* in_sizes, int n_in,
                              void* d_out, int out_size, void* d_ws, size_t ws_size,
                              hipStream_t stream) {
  (void)in_sizes; (void)n_in; (void)out_size; (void)ws_size;
  const float* z_e    = (const float*)d_in[0];
  const float* embed  = (const float*)d_in[1];
  const float* ema_cs = (const float*)d_in[2];
  const float* ema_w  = (const float*)d_in[3];
  float* out = (float*)d_out;

  char* ws = (char*)d_ws;
  float*    dw     = (float*)(ws + WS_DW);
  float*    counts = (float*)(ws + WS_COUNTS);
  float*    lossws = (float*)(ws + WS_LOSS);
  float*    enorm  = (float*)(ws + WS_ENORM);
  _Float16* eh     = (_Float16*)(ws + WS_EH);
  _Float16* el     = (_Float16*)(ws + WS_EL);

  vq_prep<<<dim3(NUM_EMB), dim3(DIM), 0, stream>>>(embed, dw, counts, lossws, enorm, eh, el);
  vq_main<<<dim3(NROWS / 128), dim3(256), LDS_TOTAL, stream>>>(
      z_e, embed, (const unsigned*)eh, (const unsigned*)el, enorm,
      out + OFF_Q, out + OFF_IDX, counts, dw, lossws);
  vq_finalize<<<dim3(1), dim3(NUM_EMB), 0, stream>>>(ema_cs, ema_w, counts, dw, lossws, out);
}
